// MultiheadAttention_32358283608276
// MI455X (gfx1250) — matrix operand order repaired and checked by reading, not compiled, not run on hardware
//
#include <hip/hip_runtime.h>

// ---------------------------------------------------------------------------
// Multi-head attention (B=2, S=2048, F=1024, H=16, D=64) for gfx1250.
// Pipeline: f32->bf16 convert, QKV GEMM (WMMA bf16), flash attention
// (WMMA bf16 + f32 online softmax), output GEMM + bias + residual.
// GEMM tiles staged with the Tensor Data Mover (tensor_load_to_lds,
// TENSORcnt); flash K/Vt chunks staged with per-lane async DMA
// (global_load_async_to_lds_b128, ASYNCcnt). Both double-buffered so the
// DMA of tile c+1 overlaps the WMMA compute of tile c.
// ---------------------------------------------------------------------------

typedef __attribute__((ext_vector_type(16))) __bf16 v16bf;
typedef __attribute__((ext_vector_type(8)))  float  v8f;
typedef __attribute__((ext_vector_type(4)))  unsigned int v4u;
typedef __attribute__((ext_vector_type(8)))  int    v8i;
typedef __attribute__((ext_vector_type(4)))  int    v4i;

constexpr int Bn = 2, Sn = 2048, Fn = 1024, Hn = 16, Dn = 64, F3n = 3072;
constexpr int SBn = Bn * Sn;   // 4096 rows

union AF { v16bf v; unsigned int u[8]; };

__device__ __forceinline__ unsigned short f2bf(float f) {
  unsigned int u = __float_as_uint(f);
  u += 0x7FFFu + ((u >> 16) & 1u);      // round-to-nearest-even
  return (unsigned short)(u >> 16);
}

__device__ __forceinline__ unsigned int lds_u32(const unsigned short* p, int idx) {
  return *reinterpret_cast<const unsigned int*>(p + idx);
}

__device__ __forceinline__ v8f v8f_zero() {
  v8f z;
#pragma unroll
  for (int i = 0; i < 8; ++i) z[i] = 0.f;
  return z;
}

__device__ __forceinline__ v8f wmma_bf16(const AF& a, const AF& b, v8f c) {
  return __builtin_amdgcn_wmma_f32_16x16x32_bf16(
      /*neg_a=*/false, a.v, /*neg_b=*/false, b.v,
      /*c_mod=*/(short)0, c, /*reuse_a=*/false, /*reuse_b=*/false);
}

// ---- gfx1250 async DMA: 16B/lane global -> LDS, tracked by ASYNCcnt -------
__device__ __forceinline__ void async_b128(const unsigned short* lds_dst,
                                           const unsigned short* gsrc) {
  unsigned int lds_addr = (unsigned int)(unsigned long long)lds_dst;  // low 32b = LDS offset
  unsigned long long gaddr = (unsigned long long)gsrc;
  asm volatile("global_load_async_to_lds_b128 %0, %1, off"
               :: "v"(lds_addr), "v"(gaddr)
               : "memory");
}
__device__ __forceinline__ void wait_async0() {
  asm volatile("s_wait_asynccnt 0x0" ::: "memory");
}

// ---- gfx1250 Tensor Data Mover: 2D bf16 tile global -> LDS (TENSORcnt) ----
#if __has_builtin(__builtin_amdgcn_tensor_load_to_lds) && \
    __has_builtin(__builtin_amdgcn_s_wait_tensorcnt)
#define USE_TDM 1
#else
#define USE_TDM 0
#endif

#if USE_TDM
// Load tile_h rows x tile_w (2B elems) from a row-major tensor whose row
// stride is `stride` elems; LDS rows padded by 4 DWORDs per 16 DWORDs
// (pad_interval=3, pad_amount=3) => LDS pitch = tile_w + 8 halves.
__device__ __forceinline__ void tdm_load_2d(unsigned int lds_addr,
                                            const void* gaddr,
                                            unsigned int stride,
                                            unsigned int tile_w,
                                            unsigned int tile_h) {
  unsigned long long ga = (unsigned long long)gaddr;
  v4u g0;
  g0.x = 1u;                                                  // count=1, user D#
  g0.y = lds_addr;                                            // LDS byte address
  g0.z = (unsigned int)(ga & 0xFFFFFFFFu);                    // global addr [31:0]
  g0.w = (unsigned int)((ga >> 32) & 0x01FFFFFFu) | (2u << 30); // [56:32] | type=2
  const unsigned int dim0 = stride;                           // tensor_dim0 (elems)
  const unsigned int dim1 = 0x100000u;                        // large (OOB never hit)
  v8i g1;
  g1[0] = (int)((1u << 16)                                    // data_size = 2B
              | (1u << 20)                                    // pad_enable
              | (3u << 22)                                    // pad_interval: 16 DW
              | (3u << 25));                                  // pad_amount:   4 DW
  g1[1] = (int)((dim0 & 0xFFFFu) << 16);                      // tensor_dim0[15:0]
  g1[2] = (int)(((dim0 >> 16) & 0xFFFFu) | ((dim1 & 0xFFFFu) << 16));
  g1[3] = (int)(((dim1 >> 16) & 0xFFFFu) | ((tile_w & 0xFFFFu) << 16)); // tile_dim0
  g1[4] = (int)(tile_h & 0xFFFFu);                            // tile_dim1; tile_dim2=0
  g1[5] = (int)stride;                                        // tensor_dim0_stride[31:0]
  g1[6] = 0;                                                  // stride hi, dim1_stride lo
  g1[7] = 0;                                                  // dim1_stride hi
  v4i z4 = {0, 0, 0, 0};
#if __clang_major__ >= 23
  v8i z8 = {0, 0, 0, 0, 0, 0, 0, 0};
  __builtin_amdgcn_tensor_load_to_lds(g0, g1, z4, z4, z8, 0);
#else
  __builtin_amdgcn_tensor_load_to_lds(g0, g1, z4, z4, 0);
#endif
}
#endif  // USE_TDM

// ---------------------------------------------------------------------------
// Convert kernels
// ---------------------------------------------------------------------------
__global__ void cvt_bf16_kernel(const float* __restrict__ src,
                                unsigned short* __restrict__ dst, int n) {
  int i = blockIdx.x * blockDim.x + threadIdx.x;
  if (i < n) dst[i] = f2bf(src[i]);
}

// dst[c*rows + r] = bf16(src[r*cols + c])   (store W transposed, [N][K])
__global__ void cvt_bf16_t_kernel(const float* __restrict__ src,
                                  unsigned short* __restrict__ dst,
                                  int rows, int cols) {
  int i = blockIdx.x * blockDim.x + threadIdx.x;
  if (i < rows * cols) {
    int r = i / cols, c = i % cols;
    dst[(size_t)c * rows + r] = f2bf(src[i]);
  }
}

// ---------------------------------------------------------------------------
// Shared 64x64-tile bf16 GEMM: A[M][K] row-major, Bt[N][K] row-major.
// 128 threads = 4 waves, each wave owns a 32x32 quadrant (2x2 WMMA subtiles).
// Double-buffered LDS (pitch 40 halves = 80B rows; 16B-aligned segments).
// ---------------------------------------------------------------------------
constexpr int GP = 40;                 // GEMM LDS pitch (halves)
constexpr int GBUF = 64 * GP;          // one LDS buffer (halves)

__device__ __forceinline__ void gemm_issue_tile(
    const unsigned short* A, int lda, const unsigned short* Bt, int ldb,
    int m0, int n0, int kc,
    unsigned short* ldsA, unsigned short* ldsB, int buf) {
#if USE_TDM
  if ((threadIdx.x >> 5) == 0) {       // wave 0 issues both tile DMAs
    tdm_load_2d((unsigned int)(unsigned long long)(ldsA + buf * GBUF),
                A + (size_t)m0 * lda + kc, (unsigned int)lda, 32u, 64u);
    tdm_load_2d((unsigned int)(unsigned long long)(ldsB + buf * GBUF),
                Bt + (size_t)n0 * ldb + kc, (unsigned int)ldb, 32u, 64u);
  }
#else
  const int tid = threadIdx.x;
#pragma unroll
  for (int it = 0; it < 2; ++it) {
    int s = tid + it * 128;            // 256 x 16B slots per matrix
    int row = s >> 2, seg = s & 3;     // 4 x 16B per 32-half row
    async_b128(ldsA + buf * GBUF + row * GP + seg * 8,
               A + (size_t)(m0 + row) * lda + kc + seg * 8);
    async_b128(ldsB + buf * GBUF + row * GP + seg * 8,
               Bt + (size_t)(n0 + row) * ldb + kc + seg * 8);
  }
#endif
}

__device__ __forceinline__ void gemm_wait_tile() {
#if USE_TDM
  if ((threadIdx.x >> 5) == 0) __builtin_amdgcn_s_wait_tensorcnt(0);
#else
  wait_async0();
#endif
}

__device__ __forceinline__ void gemm_mainloop(
    const unsigned short* __restrict__ A, int lda,
    const unsigned short* __restrict__ Bt, int ldb,
    int m0, int n0, int K,
    unsigned short* ldsA, unsigned short* ldsB, v8f acc[2][2]) {
  const int tid = threadIdx.x;
  const int lane = tid & 31, wid = tid >> 5;
  const int wm = wid >> 1, wn = wid & 1;
  const int half = lane >> 4, ln = lane & 15;

#pragma unroll
  for (int ti = 0; ti < 2; ++ti) {
#pragma unroll
    for (int tj = 0; tj < 2; ++tj) acc[ti][tj] = v8f_zero();
  }

  const int nk = K >> 5;
  gemm_issue_tile(A, lda, Bt, ldb, m0, n0, 0, ldsA, ldsB, 0);
  for (int c = 0; c < nk; ++c) {
    gemm_wait_tile();                  // my chunk-c DMAs done
    __syncthreads();                   // everyone's DMAs done; prev reads done
    if (c + 1 < nk) {
      gemm_issue_tile(A, lda, Bt, ldb, m0, n0, (c + 1) << 5, ldsA, ldsB, (c + 1) & 1);
    }
    const unsigned short* cA = ldsA + (c & 1) * GBUF;
    const unsigned short* cB = ldsB + (c & 1) * GBUF;

    AF a0, a1, b0, b1;
#pragma unroll
    for (int p = 0; p < 8; ++p) {
      int ka = 2 * p + 16 * half;                       // A-fragment k packing
      int kb = 2 * p + 16 * half;                       // B-fragment k packing
      a0.u[p] = lds_u32(cA, (wm * 32 + ln) * GP + ka);
      a1.u[p] = lds_u32(cA, (wm * 32 + 16 + ln) * GP + ka);
      b0.u[p] = lds_u32(cB, (wn * 32 + ln) * GP + kb);
      b1.u[p] = lds_u32(cB, (wn * 32 + 16 + ln) * GP + kb);
    }
    acc[0][0] = wmma_bf16(a0, b0, acc[0][0]);
    acc[0][1] = wmma_bf16(a0, b1, acc[0][1]);
    acc[1][0] = wmma_bf16(a1, b0, acc[1][0]);
    acc[1][1] = wmma_bf16(a1, b1, acc[1][1]);
  }
}

// ---------------------------------------------------------------------------
// QKV GEMM: Xbf[4096][1024] x WqkvT -> Q/K bf16 [B][H][S][D], V transposed
// as Vt bf16 [B][H][D][S] (so flash PV B-tiles stage without a transpose).
// ---------------------------------------------------------------------------
__global__ void __launch_bounds__(128) qkv_gemm_kernel(
    const unsigned short* __restrict__ Xbf, const unsigned short* __restrict__ Wt,
    const float* __restrict__ bqkv,
    unsigned short* __restrict__ Qb, unsigned short* __restrict__ Kb,
    unsigned short* __restrict__ Vt) {
  __shared__ unsigned short ldsA[2 * GBUF];
  __shared__ unsigned short ldsB[2 * GBUF];
  const int n0 = blockIdx.x * 64, m0 = blockIdx.y * 64;
  v8f acc[2][2];
  gemm_mainloop(Xbf, Fn, Wt, Fn, m0, n0, Fn, ldsA, ldsB, acc);

  const int tid = threadIdx.x;
  const int lane = tid & 31, wid = tid >> 5;
  const int wm = wid >> 1, wn = wid & 1;
  const int half = lane >> 4, ln = lane & 15;
#pragma unroll
  for (int ti = 0; ti < 2; ++ti) {
#pragma unroll
    for (int tj = 0; tj < 2; ++tj) {
      int ng = n0 + wn * 32 + tj * 16 + ln;          // 0..3071
      float bias = bqkv[ng];
      int which = ng >> 10, rem = ng & 1023;
      int hh = rem >> 6, d = rem & 63;
#pragma unroll
      for (int r = 0; r < 8; ++r) {
        int mg = m0 + wm * 32 + ti * 16 + r + 8 * half;  // 0..4095
        int bI = mg >> 11, sI = mg & 2047;
        unsigned short val = f2bf(acc[ti][tj][r] + bias);
        if (which == 0) {
          Qb[(size_t)((bI * Hn + hh) * Sn + sI) * Dn + d] = val;
        } else if (which == 1) {
          Kb[(size_t)((bI * Hn + hh) * Sn + sI) * Dn + d] = val;
        } else {
          Vt[(size_t)((bI * Hn + hh) * Dn + d) * Sn + sI] = val;  // transposed
        }
      }
    }
  }
}

// ---------------------------------------------------------------------------
// Flash attention: one WG per (b, h, 64-query tile); 4 waves x 16 q-rows each.
// K/Vt chunks double-buffered via per-lane async DMA; 4 WMMAs QK^T + online
// softmax + 4 WMMAs PV per wave per 32-key chunk.
// ---------------------------------------------------------------------------
constexpr int QP = 72;                 // ldsQ/ldsK pitch (halves), 144B rows
constexpr int VP = 40;                 // ldsVt/ldsP pitch (halves), 80B rows
constexpr int KBUF = 32 * QP;
constexpr int VBUF = 64 * VP;

__global__ void __launch_bounds__(128) flash_kernel(
    const unsigned short* __restrict__ Q, const unsigned short* __restrict__ K,
    const unsigned short* __restrict__ Vt, const unsigned char* __restrict__ pad,
    unsigned short* __restrict__ O) {
  __shared__ unsigned short ldsQ[64 * QP];     // [qrow][d]
  __shared__ unsigned short ldsK[2 * KBUF];    // [kv][d], double-buffered
  __shared__ unsigned short ldsVt[2 * VBUF];   // [d][kv], double-buffered
  __shared__ unsigned short ldsP[4][16 * VP];  // per-wave P tile [m][kv]

  const int qt = blockIdx.x, h = blockIdx.y, b = blockIdx.z;
  const int q0 = qt * 64;
  const unsigned short* Qbh  = Q  + (size_t)(b * Hn + h) * Sn * Dn;
  const unsigned short* Kbh  = K  + (size_t)(b * Hn + h) * Sn * Dn;
  const unsigned short* Vtbh = Vt + (size_t)(b * Hn + h) * Dn * Sn;

  const int tid = threadIdx.x;
  const int lane = tid & 31, wid = tid >> 5;
  const int half = lane >> 4, ln = lane & 15;

  // async-stage Q tile [64 x 64] (512 x 16B slots)
#pragma unroll
  for (int it = 0; it < 4; ++it) {
    int s = tid + it * 128;
    int row = s >> 3, seg = s & 7;
    async_b128(ldsQ + row * QP + seg * 8,
               Qbh + (size_t)(q0 + row) * Dn + seg * 8);
  }
  // async-stage chunk 0 of K [32 x 64] and Vt [64 x 32]
  const int nch = q0 / 32 + 2;         // causal: keys up to q0+63
  {
#pragma unroll
    for (int it = 0; it < 2; ++it) {
      int s = tid + it * 128;
      int krow = s >> 3, kseg = s & 7;                 // 256 slots
      async_b128(ldsK + krow * QP + kseg * 8,
                 Kbh + (size_t)krow * Dn + kseg * 8);
      int drow = s >> 2, vseg = s & 3;                 // 256 slots
      async_b128(ldsVt + drow * VP + vseg * 8,
                 Vtbh + (size_t)drow * Sn + vseg * 8);
    }
  }

  AF qa[2];
  float mrow[8], lrow[8], alpha_arr[8];
  v8f oacc[4];
#pragma unroll
  for (int r = 0; r < 8; ++r) { mrow[r] = -__builtin_inff(); lrow[r] = 0.f; }
#pragma unroll
  for (int dt = 0; dt < 4; ++dt) oacc[dt] = v8f_zero();

  for (int c = 0; c < nch; ++c) {
    wait_async0();
    __syncthreads();
    if (c == 0) {                      // Q fragments (d-chunks 0,1), loaded once
#pragma unroll
      for (int dc = 0; dc < 2; ++dc) {
#pragma unroll
        for (int p = 0; p < 8; ++p) {
          int ka = 2 * p + 16 * half + 32 * dc;
          qa[dc].u[p] = lds_u32(ldsQ, (wid * 16 + ln) * QP + ka);
        }
      }
    }
    if (c + 1 < nch) {                 // prefetch next K/Vt chunk
      const int kv0n = (c + 1) * 32;
      const int nb = (c + 1) & 1;
#pragma unroll
      for (int it = 0; it < 2; ++it) {
        int s = tid + it * 128;
        int krow = s >> 3, kseg = s & 7;
        async_b128(ldsK + nb * KBUF + krow * QP + kseg * 8,
                   Kbh + (size_t)(kv0n + krow) * Dn + kseg * 8);
        int drow = s >> 2, vseg = s & 3;
        async_b128(ldsVt + nb * VBUF + drow * VP + vseg * 8,
                   Vtbh + (size_t)drow * Sn + kv0n + vseg * 8);
      }
    }
    const int kv0 = c * 32;
    const unsigned short* cK = ldsK + (c & 1) * KBUF;
    const unsigned short* cV = ldsVt + (c & 1) * VBUF;

    // scores S = Q . K^T
    v8f sc[2];
#pragma unroll
    for (int nt = 0; nt < 2; ++nt) {
      sc[nt] = v8f_zero();
#pragma unroll
      for (int dc = 0; dc < 2; ++dc) {
        AF kb;
#pragma unroll
        for (int p = 0; p < 8; ++p) {
          kb.u[p] = lds_u32(cK, (nt * 16 + ln) * QP + 2 * p + 16 * half + 32 * dc);
        }
        sc[nt] = wmma_bf16(qa[dc], kb, sc[nt]);
      }
    }

    // scale + causal/padding mask + online softmax
    const int kg0 = kv0 + ln, kg1 = kv0 + 16 + ln;
    const bool pm0 = pad[b * Sn + kg0] != 0;
    const bool pm1 = pad[b * Sn + kg1] != 0;
#pragma unroll
    for (int r = 0; r < 8; ++r) {
      int qg = q0 + wid * 16 + r + 8 * half;
      float s0 = sc[0][r] * 0.125f;                  // 1/sqrt(64)
      float s1 = sc[1][r] * 0.125f;
      if (pm0 || kg0 > qg) s0 = -__builtin_inff();
      if (pm1 || kg1 > qg) s1 = -__builtin_inff();
      float v = fmaxf(s0, s1);
      v = fmaxf(v, __shfl_xor(v, 1));
      v = fmaxf(v, __shfl_xor(v, 2));
      v = fmaxf(v, __shfl_xor(v, 4));
      v = fmaxf(v, __shfl_xor(v, 8));
      float mnew = fmaxf(mrow[r], v);
      float alpha, p0, p1;
      if (mnew == -__builtin_inff()) {               // fully-masked row so far
        alpha = 1.f; p0 = 0.f; p1 = 0.f;
      } else {
        alpha = __expf(mrow[r] - mnew);
        p0 = __expf(s0 - mnew);
        p1 = __expf(s1 - mnew);
      }
      float rs = p0 + p1;
      rs += __shfl_xor(rs, 1);
      rs += __shfl_xor(rs, 2);
      rs += __shfl_xor(rs, 4);
      rs += __shfl_xor(rs, 8);
      lrow[r] = lrow[r] * alpha + rs;
      mrow[r] = mnew;
      alpha_arr[r] = alpha;
      sc[0][r] = p0; sc[1][r] = p1;
    }

    // rescale O accumulators
#pragma unroll
    for (int dt = 0; dt < 4; ++dt) {
#pragma unroll
      for (int r = 0; r < 8; ++r) oacc[dt][r] *= alpha_arr[r];
    }

    // transpose P (C-layout -> A-layout) through per-wave LDS
    unsigned short* Pw = &ldsP[wid][0];
#pragma unroll
    for (int r = 0; r < 8; ++r) {
      Pw[(r + 8 * half) * VP + ln]      = f2bf(sc[0][r]);
      Pw[(r + 8 * half) * VP + 16 + ln] = f2bf(sc[1][r]);
    }
    AF pa;
#pragma unroll
    for (int p = 0; p < 8; ++p) {
      pa.u[p] = lds_u32(Pw, ln * VP + 2 * p + 16 * half);
    }
    // O += P . V
#pragma unroll
    for (int dt = 0; dt < 4; ++dt) {
      AF vb;
#pragma unroll
      for (int p = 0; p < 8; ++p) {
        vb.u[p] = lds_u32(cV, (dt * 16 + ln) * VP + 2 * p + 16 * half);
      }
      oacc[dt] = wmma_bf16(pa, vb, oacc[dt]);
    }
  }

  // finalize: divide by l (0 if row fully masked) and store merged-head bf16
  float inv[8];
#pragma unroll
  for (int r = 0; r < 8; ++r) inv[r] = (lrow[r] > 0.f) ? (1.f / lrow[r]) : 0.f;
#pragma unroll
  for (int dt = 0; dt < 4; ++dt) {
#pragma unroll
    for (int r = 0; r < 8; ++r) {
      int qg = q0 + wid * 16 + r + 8 * half;
      int col = h * Dn + dt * 16 + ln;
      O[(size_t)(b * Sn + qg) * Fn + col] = f2bf(oacc[dt][r] * inv[r]);
    }
  }
}

// ---------------------------------------------------------------------------
// Output GEMM: out = x + Obf @ Wout + b_out   (f32 output)
// ---------------------------------------------------------------------------
__global__ void __launch_bounds__(128) out_gemm_kernel(
    const unsigned short* __restrict__ Obf, const unsigned short* __restrict__ Wt,
    const float* __restrict__ bout, const float* __restrict__ x,
    float* __restrict__ out) {
  __shared__ unsigned short ldsA[2 * GBUF];
  __shared__ unsigned short ldsB[2 * GBUF];
  const int n0 = blockIdx.x * 64, m0 = blockIdx.y * 64;
  v8f acc[2][2];
  gemm_mainloop(Obf, Fn, Wt, Fn, m0, n0, Fn, ldsA, ldsB, acc);

  const int tid = threadIdx.x;
  const int lane = tid & 31, wid = tid >> 5;
  const int wm = wid >> 1, wn = wid & 1;
  const int half = lane >> 4, ln = lane & 15;
#pragma unroll
  for (int ti = 0; ti < 2; ++ti) {
#pragma unroll
    for (int tj = 0; tj < 2; ++tj) {
      int ng = n0 + wn * 32 + tj * 16 + ln;
      float bias = bout[ng];
#pragma unroll
      for (int r = 0; r < 8; ++r) {
        int mg = m0 + wm * 32 + ti * 16 + r + 8 * half;
        size_t idx = (size_t)mg * Fn + ng;
        out[idx] = x[idx] + bias + acc[ti][tj][r];
      }
    }
  }
}

// ---------------------------------------------------------------------------
// Host launcher
// ---------------------------------------------------------------------------
extern "C" void kernel_launch(void* const* d_in, const int* in_sizes, int n_in,
                              void* d_out, int out_size, void* d_ws, size_t ws_size,
                              hipStream_t stream) {
  const float*         x    = (const float*)d_in[0];
  const unsigned char* pad  = (const unsigned char*)d_in[1];  // bool mask, 1B/elem
  const float*         wqkv = (const float*)d_in[2];
  const float*         bqkv = (const float*)d_in[3];
  const float*         wout = (const float*)d_in[4];
  const float*         bout = (const float*)d_in[5];
  float*               out  = (float*)d_out;

  char* ws = (char*)d_ws;
  // Xbf is consumed by qkv_gemm before flash writes Obf -> alias them.
  unsigned short* Xbf   = (unsigned short*)(ws + 0);          //  8 MiB (aliased w/ Obf)
  unsigned short* Obf   = (unsigned short*)(ws + 0);
  unsigned short* WqkvT = (unsigned short*)(ws + 8388608);    //  6 MiB
  unsigned short* WoutT = (unsigned short*)(ws + 14680064);   //  2 MiB
  unsigned short* Qb    = (unsigned short*)(ws + 16777216);   //  8 MiB
  unsigned short* Kb    = (unsigned short*)(ws + 25165824);   //  8 MiB
  unsigned short* Vt    = (unsigned short*)(ws + 33554432);   //  8 MiB  (total 40 MiB)

  cvt_bf16_kernel<<<(SBn * Fn + 255) / 256, 256, 0, stream>>>(x, Xbf, SBn * Fn);
  cvt_bf16_t_kernel<<<(Fn * F3n + 255) / 256, 256, 0, stream>>>(wqkv, WqkvT, Fn, F3n);
  cvt_bf16_t_kernel<<<(Fn * Fn + 255) / 256, 256, 0, stream>>>(wout, WoutT, Fn, Fn);

  qkv_gemm_kernel<<<dim3(F3n / 64, SBn / 64), 128, 0, stream>>>(Xbf, WqkvT, bqkv, Qb, Kb, Vt);
  flash_kernel<<<dim3(Sn / 64, Hn, Bn), 128, 0, stream>>>(Qb, Kb, Vt, pad, Obf);
  out_gemm_kernel<<<dim3(Fn / 64, SBn / 64), 128, 0, stream>>>(Obf, WoutT, bout, x, out);
}
